// SwinTransformer_32650341384499
// MI455X (gfx1250) — matrix operand order repaired and checked by reading, not compiled, not run on hardware
//
#include <hip/hip_runtime.h>
#include <hip/hip_bf16.h>

typedef __bf16 bf16_t;
typedef __attribute__((ext_vector_type(16))) __bf16 v16bf;
typedef __attribute__((ext_vector_type(8)))  float  v8f;
typedef __attribute__((ext_vector_type(4)))  int    v4i;

#define B_   8
#define NW_  64
#define P_   160
#define E_   512
#define H_   8
#define D_   64

// ----------------------------------------------------- async global->LDS DMA
// CDNA5 GLOBAL_LOAD_ASYNC_TO_LDS_B128 (ASYNCcnt-tracked); falls back to a
// sync register round-trip if this toolchain doesn't declare the builtin.

#if __has_builtin(__builtin_amdgcn_global_load_async_to_lds_b128)
#define HAVE_ASYNC_LDS 1
#endif

typedef __attribute__((address_space(1))) v4i* g4p;   // global int4*
typedef __attribute__((address_space(3))) v4i* l4p;   // LDS int4*

__device__ __forceinline__ void g2l_b128(const void* g, void* l) {
#ifdef HAVE_ASYNC_LDS
  // Flat LDS address [31:0] is the LDS offset (ISA 10.2 aperture rules).
  __builtin_amdgcn_global_load_async_to_lds_b128(
      (g4p)(unsigned long long)g,
      (l4p)(unsigned)(unsigned long long)l, /*offset=*/0, /*cpol=*/0);
#else
  *(uint4*)l = *(const uint4*)g;
#endif
}

__device__ __forceinline__ void async_wait0() {
#ifdef HAVE_ASYNC_LDS
#if __has_builtin(__builtin_amdgcn_s_wait_asynccnt)
  __builtin_amdgcn_s_wait_asynccnt(0);
#else
  asm volatile("s_wait_asynccnt 0x0" ::: "memory");
#endif
#endif
}

// ---------------------------------------------------------------- WMMA helpers

__device__ __forceinline__ v8f wmma_bf16(v16bf a, v16bf b, v8f c) {
  // D = A(16x32 bf16) * B(32x16 bf16) + C(16x16 f32)
  return __builtin_amdgcn_wmma_f32_16x16x32_bf16(
      /*neg_a=*/false, a, /*neg_b=*/false, b,
      /*c_mod=*/(short)0, c, /*reuse_a=*/false, /*reuse_b=*/false);
}

// A-matrix 16x32 bf16 fragment from row-major LDS tile (ld = row stride, elems).
// lanes 0-15 -> M=lane, K={0..7,16..23}; lanes 16-31 -> M=lane-16, K={8..15,24..31}
__device__ __forceinline__ v16bf load_afrag(const bf16_t* s, int ld, int m0,
                                            int k0, int lane) {
  const int m  = m0 + (lane & 15);
  const int kb = (lane >> 4) << 3;            // 0 or 8
  const unsigned* row = (const unsigned*)(s + m * ld);
  union { unsigned u[8]; v16bf v; } r;
  const int c0 = (k0 + kb) >> 1;              // dword index
  const int c1 = (k0 + 16 + kb) >> 1;
#pragma unroll
  for (int i = 0; i < 4; ++i) r.u[i] = row[c0 + i];
#pragma unroll
  for (int i = 0; i < 4; ++i) r.u[4 + i] = row[c1 + i];
  return r.v;
}

// B-matrix 32x16 bf16 fragment from "B-transposed row-major" storage (rows = n,
// cols = k). lanes 0-15 -> n=lane, K={0..7,16..23}; lanes 16-31 -> n=lane-16, K={8..15,24..31}.
__device__ __forceinline__ v16bf load_bfrag(const bf16_t* s, int ld, int n0,
                                            int k0, int lane) {
  const int n  = n0 + (lane & 15);
  const int kb = (lane >> 4) << 3;            // 0 or 8
  const unsigned* row = (const unsigned*)(s + n * ld);
  union { unsigned u[8]; v16bf v; } r;
  const int c0 = (k0 + kb) >> 1;
  const int c1 = (k0 + 16 + kb) >> 1;
#pragma unroll
  for (int i = 0; i < 4; ++i) r.u[i] = row[c0 + i];
#pragma unroll
  for (int i = 0; i < 4; ++i) r.u[4 + i] = row[c1 + i];
  return r.v;
}

// ------------------------------------------------------------ fp32 -> bf16

__global__ void cvt_bf16_kernel(const float* __restrict__ src,
                                bf16_t* __restrict__ dst, int n) {
  int i = (blockIdx.x * blockDim.x + threadIdx.x) * 4;
  if (i + 3 < n) {
    float4 v = *(const float4*)(src + i);
    bf16_t o[4] = {(bf16_t)v.x, (bf16_t)v.y, (bf16_t)v.z, (bf16_t)v.w};
    *(uint2*)(dst + i) = *(const uint2*)o;
  } else {
    for (int j = i; j < n; ++j) dst[j] = (bf16_t)src[j];
  }
}

// --------------------------------------------- projection GEMM: Y = X * W^T + b
// X: [81920 x 512] bf16. W: [512 x 512] bf16 (W[f][e]) -> rows are k-contiguous.
// Double-buffered async-LDS pipeline; output head-split [B,H,NW,P,D] bf16.

__global__ __launch_bounds__(256) void proj_gemm_kernel(
    const bf16_t* __restrict__ X, const bf16_t* __restrict__ W,
    const float* __restrict__ bias, bf16_t* __restrict__ Y, float outScale) {
  __shared__ bf16_t Xs[2][128 * 40];   // 128 rows x 32 k, pad to 40
  __shared__ bf16_t Ws[2][64 * 40];    // 64 n-rows x 32 k, pad to 40

  const int tid  = threadIdx.x;
  const int lane = tid & 31;
  const int wv   = tid >> 5;                  // wave id = M-tile 0..7
  const int m0   = blockIdx.x * 128;
  const int n0   = blockIdx.y * 64;

  auto stage = [&](int buf, int ks) {
    const int k0 = ks * 32;
#pragma unroll
    for (int i = 0; i < 2; ++i) {             // 128x32 of X (async DMA)
      int idx = tid + i * 256;
      int row = idx >> 2, seg = (idx & 3) * 8;
      g2l_b128(X + (size_t)(m0 + row) * E_ + k0 + seg, &Xs[buf][row * 40 + seg]);
    }
    {                                         // 64x32 of W (async DMA)
      int row = tid >> 2, seg = (tid & 3) * 8;
      g2l_b128(W + (size_t)(n0 + row) * E_ + k0 + seg, &Ws[buf][row * 40 + seg]);
    }
  };

  v8f acc[4] = {};

  stage(0, 0);
  async_wait0();
  __syncthreads();

  for (int ks = 0; ks < 16; ++ks) {
    const int cur = ks & 1;
    if (ks + 1 < 16) stage(cur ^ 1, ks + 1);      // overlap DMA with WMMA
    if (ks + 2 < 16)                              // L2 prefetch two tiles ahead
      __builtin_prefetch(X + (size_t)(m0 + (tid >> 1)) * E_ + (ks + 2) * 32, 0, 3);

    v16bf a = load_afrag(Xs[cur], 40, wv * 16, 0, lane);
#pragma unroll
    for (int nt = 0; nt < 4; ++nt) {
      v16bf b = load_bfrag(Ws[cur], 40, nt * 16, 0, lane);
      acc[nt] = wmma_bf16(a, b, acc[nt]);
    }
    async_wait0();
    __syncthreads();
  }

  // epilogue: +bias, *scale, scatter to [B,H,NW,P,D] bf16
  const int mhalf = (lane >> 4) << 3;
#pragma unroll
  for (int nt = 0; nt < 4; ++nt) {
    const int f  = n0 + nt * 16 + (lane & 15);
    const float bv = bias[f];
    const int h = f >> 6, d = f & 63;
#pragma unroll
    for (int r = 0; r < 8; ++r) {
      const int token = m0 + wv * 16 + r + mhalf;
      const int b_  = token / (NW_ * P_);
      const int rem = token - b_ * (NW_ * P_);
      const int w_  = rem / P_;
      const int p_  = rem - w_ * P_;
      const size_t o = ((((size_t)b_ * H_ + h) * NW_ + w_) * P_ + p_) * D_ + d;
      Y[o] = (bf16_t)((acc[nt][r] + bv) * outScale);
    }
  }
}

// ----------------------------------------------- windowed attention per (b,h,w)
// 320 threads = 10 waves; wave mt owns Q-rows [mt*16, mt*16+16).

__global__ __launch_bounds__(320) void swin_attn_kernel(
    const bf16_t* __restrict__ Q, const bf16_t* __restrict__ K,
    const bf16_t* __restrict__ V, const float* __restrict__ pos_bias,
    float* __restrict__ out) {
  __shared__ bf16_t Qs[P_ * D_];        // 160 x 64
  __shared__ bf16_t Ks[P_ * D_];        // 160 x 64
  __shared__ bf16_t Vt[D_ * 168];       // V transposed: [d][p], pad 168
  __shared__ bf16_t Pa[P_ * 168];       // softmax probs bf16, pad 168
  __shared__ float  pb[19 * 31];        // relative position bias table

  const int tid  = threadIdx.x;
  const int lane = tid & 31;
  const int mt   = tid >> 5;            // 0..9

  const int idx = blockIdx.x;           // enumerates (b,h,w) in Q/K/V order
  const int w   = idx % NW_;
  const int h   = (idx / NW_) % H_;
  const int b   = idx / (H_ * NW_);
  const size_t base = (size_t)idx * (P_ * D_);

  // ---- stage Q, K via async LDS-DMA; V transposed via ds scatter; bias table
  {
#pragma unroll
    for (int i = 0; i < 4; ++i) {
      int e = (tid + i * 320) * 8;      // 16B chunks, 20KB each matrix
      g2l_b128(Q + base + e, Qs + e);
      g2l_b128(K + base + e, Ks + e);
    }
    const unsigned* gv = (const unsigned*)(V + base);
#pragma unroll
    for (int i = 0; i < 16; ++i) {
      int e = tid + i * 320;            // 5120 dword pairs
      int p = e >> 5;                   // 32 dwords per 64-elem row
      int d = (e & 31) * 2;
      union { unsigned uu; bf16_t hh[2]; } cv; cv.uu = gv[e];
      Vt[d * 168 + p]       = cv.hh[0];
      Vt[(d + 1) * 168 + p] = cv.hh[1];
    }
    for (int i = tid; i < 19 * 31; i += 320) pb[i] = pos_bias[i];
  }
  async_wait0();
  __syncthreads();

  // ---- S = Q K^T   (10 N-tiles x 2 k-steps = 20 WMMA per wave)
  v16bf aq0 = load_afrag(Qs, D_, mt * 16, 0, lane);
  v16bf aq1 = load_afrag(Qs, D_, mt * 16, 32, lane);
  v8f acc[10] = {};
#pragma unroll
  for (int j = 0; j < 10; ++j) {
    v16bf b0 = load_bfrag(Ks, D_, j * 16, 0, lane);
    v16bf b1 = load_bfrag(Ks, D_, j * 16, 32, lane);
    acc[j] = wmma_bf16(aq0, b0, acc[j]);
    acc[j] = wmma_bf16(aq1, b1, acc[j]);
  }

  // ---- scale + rel-pos bias + analytic shift mask + row softmax
  const int wy = w >> 3, wx = w & 7;
  const int nlo = lane & 15;
  const float NEG = -__builtin_inff();
#pragma unroll
  for (int r = 0; r < 8; ++r) {
    const int m  = mt * 16 + r + ((lane >> 4) << 3);
    const int vm = m >> 4, hm = m & 15;
    float mx = NEG;
#pragma unroll
    for (int j = 0; j < 10; ++j) {
      const int n  = j * 16 + nlo;
      const int vn = n >> 4, hn = n & 15;
      const bool msk = ((wy == 7) && ((vm < 5) != (vn < 5))) ||
                       ((wx == 7) && ((hm < 8) != (hn < 8)));
      float lg = acc[j][r] * 0.125f +
                 pb[((vm - vn + 19) % 19) * 31 + ((hm - hn + 31) % 31)];
      lg = msk ? NEG : lg;
      acc[j][r] = lg;
      mx = fmaxf(mx, lg);
    }
    mx = fmaxf(mx, __shfl_xor(mx, 1, 32));
    mx = fmaxf(mx, __shfl_xor(mx, 2, 32));
    mx = fmaxf(mx, __shfl_xor(mx, 4, 32));
    mx = fmaxf(mx, __shfl_xor(mx, 8, 32));
    float sum = 0.f;
#pragma unroll
    for (int j = 0; j < 10; ++j) {
      float e = __expf(acc[j][r] - mx);
      acc[j][r] = e;
      sum += e;
    }
    sum += __shfl_xor(sum, 1, 32);
    sum += __shfl_xor(sum, 2, 32);
    sum += __shfl_xor(sum, 4, 32);
    sum += __shfl_xor(sum, 8, 32);
    const float inv = 1.0f / sum;
#pragma unroll
    for (int j = 0; j < 10; ++j)
      Pa[m * 168 + j * 16 + nlo] = (bf16_t)(acc[j][r] * inv);
  }
  __syncthreads();

  // ---- O = P V   (5 k-steps x 4 N-tiles = 20 WMMA per wave)
  v8f oacc[4] = {};
#pragma unroll
  for (int ks = 0; ks < 5; ++ks) {
    v16bf a = load_afrag(Pa, 168, mt * 16, ks * 32, lane);
#pragma unroll
    for (int nt = 0; nt < 4; ++nt) {
      v16bf bb = load_bfrag(Vt, 168, nt * 16, ks * 32, lane);
      oacc[nt] = wmma_bf16(a, bb, oacc[nt]);
    }
  }

  // ---- scatter fp32 to [B, NW, P, E]
#pragma unroll
  for (int nt = 0; nt < 4; ++nt) {
    const int d = nt * 16 + nlo;
#pragma unroll
    for (int r = 0; r < 8; ++r) {
      const int p = mt * 16 + r + ((lane >> 4) << 3);
      const size_t o = (((size_t)b * NW_ + w) * P_ + p) * E_ + h * D_ + d;
      out[o] = oacc[nt][r];
    }
  }
}

// ---------------------------------------------------------------- launch

extern "C" void kernel_launch(void* const* d_in, const int* in_sizes, int n_in,
                              void* d_out, int out_size, void* d_ws, size_t ws_size,
                              hipStream_t stream) {
  const float* patches  = (const float*)d_in[0];
  const float* wq       = (const float*)d_in[1];
  const float* bq       = (const float*)d_in[2];
  const float* wk       = (const float*)d_in[3];
  const float* bk       = (const float*)d_in[4];
  const float* wv       = (const float*)d_in[5];
  const float* bv       = (const float*)d_in[6];
  const float* pos_bias = (const float*)d_in[7];
  // d_in[8] = mask, recomputed analytically in-kernel.
  float* out = (float*)d_out;
  char*  ws  = (char*)d_ws;

  const size_t N_TOK  = (size_t)B_ * NW_ * P_;   // 81920
  const size_t X_ELEM = N_TOK * E_;              // 41,943,040
  const size_t W_ELEM = (size_t)E_ * E_;         // 262,144
  const size_t QKV_EL = X_ELEM;                  // [B,H,NW,P,D] flat

  bf16_t* Xbf = (bf16_t*)ws;
  bf16_t* Wqb = Xbf + X_ELEM;
  bf16_t* Wkb = Wqb + W_ELEM;
  bf16_t* Wvb = Wkb + W_ELEM;
  bf16_t* Qbf = Wvb + W_ELEM;
  bf16_t* Kbf = Qbf + QKV_EL;
  bf16_t* Vbf = Kbf + QKV_EL;

  // 1) fp32 -> bf16 conversions (X stays resident in the 192MB L2)
  cvt_bf16_kernel<<<(int)(X_ELEM / 1024), 256, 0, stream>>>(patches, Xbf, (int)X_ELEM);
  cvt_bf16_kernel<<<(int)(W_ELEM / 1024), 256, 0, stream>>>(wq, Wqb, (int)W_ELEM);
  cvt_bf16_kernel<<<(int)(W_ELEM / 1024), 256, 0, stream>>>(wk, Wkb, (int)W_ELEM);
  cvt_bf16_kernel<<<(int)(W_ELEM / 1024), 256, 0, stream>>>(wv, Wvb, (int)W_ELEM);

  // 2) QKV projections (Q pre-scaled by 1/sqrt(D) = 0.125)
  dim3 g((unsigned)(N_TOK / 128), E_ / 64);
  proj_gemm_kernel<<<g, 256, 0, stream>>>(Xbf, Wqb, bq, Qbf, 0.125f);
  proj_gemm_kernel<<<g, 256, 0, stream>>>(Xbf, Wkb, bk, Kbf, 1.0f);
  proj_gemm_kernel<<<g, 256, 0, stream>>>(Xbf, Wvb, bv, Vbf, 1.0f);

  // 3) windowed attention, one block per (b,h,w)
  swin_attn_kernel<<<B_ * H_ * NW_, 320, 0, stream>>>(Qbf, Kbf, Vbf, pos_bias, out);
}
